// CausalSelfAttention_78305843741196
// MI455X (gfx1250) — matrix operand order repaired and checked by reading, not compiled, not run on hardware
//
#include <hip/hip_runtime.h>
#include <cstdint>

#define B_    2
#define S_    2048
#define D_    1024
#define H_    16
#define HD_   64
#define M_TOT (B_ * S_)   // 4096

typedef unsigned short u16;
typedef __attribute__((ext_vector_type(4)))  unsigned int u32x4;
typedef __attribute__((ext_vector_type(4)))  unsigned int v4u;
typedef __attribute__((ext_vector_type(4)))  int          i32x4;
typedef __attribute__((ext_vector_type(8)))  int          i32x8;
typedef __attribute__((ext_vector_type(8)))  float        v8f;
typedef __attribute__((ext_vector_type(16))) __bf16       v16bf;

#define HAVE_TDM __has_builtin(__builtin_amdgcn_tensor_load_to_lds)

union BFrag { v16bf v; v4u q[2]; u16 u[16]; };

__device__ __forceinline__ u16 f2bf(float f) {
  union { float f; uint32_t u; } c; c.f = f;
  uint32_t u = c.u;
  return (u16)((u + 0x7FFFu + ((u >> 16) & 1u)) >> 16);  // round-to-nearest-even
}

__device__ __forceinline__ v8f wmma_bf16(const BFrag& a, const BFrag& b, v8f c) {
  return __builtin_amdgcn_wmma_f32_16x16x32_bf16(false, a.v, false, b.v,
                                                 (short)0, c, false, false);
}

// B fragment (32x16, bf16) from WT stored [N][K]: two 16B loads per lane (k = hw*8 + 0..7, 16 + hw*8 + 0..7).
__device__ __forceinline__ void loadB(BFrag& f, const u16* __restrict__ WT,
                                      int nBase, int l16, int hw, int t, int k0) {
  const u16* bp = WT + (size_t)(nBase + t*16 + l16) * D_ + k0 + hw*8;
  f.q[0] = *(const v4u*)bp;                    // k = hw*8 + 0..7
  f.q[1] = *(const v4u*)(bp + 16);             // k = 16 + hw*8 + 0..7
}

// -------------------------------------------------------------- TDM helper
// Async-DMA a 64-row x 32-col bf16 tile (row stride D_) from global into LDS.
__device__ __forceinline__ void tdm_load_a64x32(uint32_t lds_off, const u16* gptr) {
#if HAVE_TDM
  uint64_t ga = (uint64_t)(uintptr_t)gptr;
  u32x4 g0;
  g0[0] = 1u;                                               // count=1, user mode
  g0[1] = lds_off;                                          // lds_addr (bytes)
  g0[2] = (uint32_t)(ga & 0xFFFFFFFFu);                     // global_addr[31:0]
  g0[3] = (uint32_t)((ga >> 32) & 0x1FFFFFFu) | (2u << 30); // ga[56:32] | type=2
  i32x8 g1;
  g1[0] = (int)(1u << 16);                 // workgroup_mask=0, data_size=1 (2B)
  g1[1] = (int)((uint32_t)D_ << 16);       // tensor_dim0[15:0] = 1024
  g1[2] = (int)((uint32_t)M_TOT << 16);    // dim0 hi=0 | tensor_dim1 lo = 4096
  g1[3] = (int)(32u << 16);                // dim1 hi=0 | tile_dim0 = 32
  g1[4] = 64;                              // tile_dim1 = 64, tile_dim2 = 0
  g1[5] = D_;                              // tensor_dim0_stride lo = 1024
  g1[6] = 0;
  g1[7] = 0;
  i32x4 z4 = {0, 0, 0, 0};
#if __clang_major__ >= 23
  i32x8 z8 = {0, 0, 0, 0, 0, 0, 0, 0};
  __builtin_amdgcn_tensor_load_to_lds(g0, g1, z4, z4, z8, 0);
#else
  __builtin_amdgcn_tensor_load_to_lds(g0, g1, z4, z4, 0);
#endif
#endif
}

// ---------------------------------------------------------------- converts
__global__ void cvt_x_kernel(const float* __restrict__ x, u16* __restrict__ xb, int n) {
  int i = blockIdx.x * blockDim.x + threadIdx.x;
  if (i < n) xb[i] = f2bf(x[i]);
}

// WT[n][k] = bf16(W[k][n]); coalesced reads, strided stores (one-time transform)
__global__ void cvt_wt_kernel(const float* __restrict__ W, u16* __restrict__ WT) {
  int idx = blockIdx.x * blockDim.x + threadIdx.x; // = k*1024 + n
  int k = idx >> 10, n = idx & 1023;
  WT[(size_t)n * D_ + k] = f2bf(W[idx]);
}

// ------------------------------------------------------- 64x64 GEMM engine
// One wave computes a 64(M)x64(N) C tile of  A(bf16 MxK, row-major) @ WT^T
// (WT stored [N][K] bf16).  A tile is TDM-staged into double-buffered LDS;
// B fragments are software-pipelined one (t,k)-step ahead.
__device__ __forceinline__ void gemm64_compute(const u16* __restrict__ Ab,
                                               const u16* __restrict__ WT,
                                               int mBase, int nBase, int lane,
                                               u16* atile0, u16* atile1,
                                               v8f acc[4][4]) {
  const int hw = lane >> 4, l16 = lane & 15;
  const v8f z8 = {0,0,0,0,0,0,0,0};
#pragma unroll
  for (int s = 0; s < 4; ++s)
#pragma unroll
    for (int t = 0; t < 4; ++t) acc[s][t] = z8;

  BFrag bcur;
  loadB(bcur, WT, nBase, l16, hw, 0, 0);       // pipeline prologue (t=0,k=0)

#if HAVE_TDM
  const u16* ag = Ab + (size_t)mBase * D_;
  const uint32_t lds0 = (uint32_t)(uintptr_t)atile0;
  const uint32_t lds1 = (uint32_t)(uintptr_t)atile1;
  tdm_load_a64x32(lds0, ag);                   // prologue: k0 = 0 tile
  int buf = 0;
  for (int k0 = 0; k0 < D_; k0 += 32, buf ^= 1) {
    if (k0 + 32 < D_) {
      tdm_load_a64x32(buf ? lds0 : lds1, ag + (k0 + 32));    // prefetch next
      __asm__ volatile("s_wait_tensorcnt 0x1" ::: "memory"); // current ready
    } else {
      __asm__ volatile("s_wait_tensorcnt 0x0" ::: "memory");
    }
    const u16* at = buf ? atile1 : atile0;
    BFrag a[4];
#pragma unroll
    for (int s = 0; s < 4; ++s) {              // A frag rows: s*16 + l16
      const u16* ap = at + (s*16 + l16) * 32 + hw*8;
      a[s].q[0] = *(const v4u*)ap;             // k = hw*8 + 0..7
      a[s].q[1] = *(const v4u*)(ap + 16);      // k = 16 + hw*8 + 0..7
    }
#pragma unroll
    for (int t = 0; t < 4; ++t) {
      BFrag bnext = bcur;
      const int tn = (t + 1) & 3;
      const int kn = (t == 3) ? (k0 + 32) : k0;
      if (kn < D_) loadB(bnext, WT, nBase, l16, hw, tn, kn); // issue ahead
#pragma unroll
      for (int s = 0; s < 4; ++s)
        acc[s][t] = wmma_bf16(a[s], bcur, acc[s][t]);
      bcur = bnext;
    }
  }
#else
  (void)atile0; (void)atile1;
  for (int k0 = 0; k0 < D_; k0 += 32) {
    BFrag a[4];
#pragma unroll
    for (int s = 0; s < 4; ++s) {
      const u16* ap = Ab + (size_t)(mBase + s*16 + l16) * D_ + k0 + hw*8;
      a[s].q[0] = *(const v4u*)ap;
      a[s].q[1] = *(const v4u*)(ap + 16);
    }
#pragma unroll
    for (int t = 0; t < 4; ++t) {
      BFrag bnext = bcur;
      const int tn = (t + 1) & 3;
      const int kn = (t == 3) ? (k0 + 32) : k0;
      if (kn < D_) loadB(bnext, WT, nBase, l16, hw, tn, kn);
#pragma unroll
      for (int s = 0; s < 4; ++s)
        acc[s][t] = wmma_bf16(a[s], bcur, acc[s][t]);
      bcur = bnext;
    }
  }
#endif
}

// ---------------------------------------------------------------- QKV GEMM
__global__ void __launch_bounds__(32)
gemm_qkv_kernel(const u16* __restrict__ xb, const u16* __restrict__ WqT,
                const u16* __restrict__ WkT, const u16* __restrict__ WvT,
                u16* __restrict__ Qb, u16* __restrict__ Kb, u16* __restrict__ Vt) {
  __shared__ __align__(16) u16 atile[2][64 * 32];
  const int lane = threadIdx.x;
  const int hw = lane >> 4, l16 = lane & 15;
  const int mBase = blockIdx.x * 64;
  const int nBase = blockIdx.y * 64;
  const int which = blockIdx.z;  // 0=Q 1=K 2=V
  const u16* WT = (which == 0) ? WqT : ((which == 1) ? WkT : WvT);

  v8f acc[4][4];
  gemm64_compute(xb, WT, mBase, nBase, lane, &atile[0][0], &atile[1][0], acc);

#pragma unroll
  for (int s = 0; s < 4; ++s)
#pragma unroll
    for (int t = 0; t < 4; ++t)
#pragma unroll
      for (int r = 0; r < 8; ++r) {
        int row = mBase + s*16 + hw*8 + r;
        int col = nBase + t*16 + l16;
        u16 hval = f2bf(acc[s][t][r]);
        if (which == 0)      Qb[(size_t)row * D_ + col] = hval;
        else if (which == 1) Kb[(size_t)row * D_ + col] = hval;
        else                 Vt[(size_t)col * M_TOT + row] = hval;  // V transposed
      }
}

// ---------------------------------------------------------------- attention
// One wave per (b, h, 16-query-row tile). Flash-attention over 32-key blocks.
__global__ void __launch_bounds__(32)
attn_kernel(const u16* __restrict__ Qb, const u16* __restrict__ Kb,
            const u16* __restrict__ Vt, u16* __restrict__ Cmb) {
  const int lane = threadIdx.x;
  const int hw = lane >> 4, l16 = lane & 15;
  const int qtile = blockIdx.x;   // 0..127
  const int h     = blockIdx.y;   // 0..15
  const int b     = blockIdx.z;   // 0..1
  const int rowBase = b * S_;
  const int qRow0   = qtile * 16;

  __shared__ __align__(16) u16 pstage[16 * 32];

  // Q fragments: 16x64 as two 16x32 A-fragments
  BFrag aQ[2];
  {
    const u16* qrow = Qb + (size_t)(rowBase + qRow0 + l16) * D_ + h * HD_;
#pragma unroll
    for (int dc = 0; dc < 2; ++dc) {
      const u16* p = qrow + dc*32 + hw*8;
      aQ[dc].q[0] = *(const v4u*)p;
      aQ[dc].q[1] = *(const v4u*)(p + 16);
    }
  }

  const v8f z8 = {0,0,0,0,0,0,0,0};
  float m[8], lsum[8];
  v8f o[4];
#pragma unroll
  for (int r = 0; r < 8; ++r) { m[r] = -__builtin_inff(); lsum[r] = 0.f; }
#pragma unroll
  for (int t = 0; t < 4; ++t) o[t] = z8;

  const int nkb = (qRow0 + 15) / 32 + 1;
  for (int kb = 0; kb < nkb; ++kb) {
    const int kRow0 = kb * 32;

    // prefetch next key/value block (global_prefetch_b8)
    if (kb + 1 < nkb) {
      __builtin_prefetch(Kb + (size_t)(rowBase + kRow0 + 32 + lane) * D_ + h*HD_, 0, 0);
      __builtin_prefetch(Vt + (size_t)(h*HD_ + lane) * M_TOT + rowBase + kRow0 + 32, 0, 0);
    }

    // S = Q @ K^T  (16 queries x 32 keys)
    v8f sfr[2] = {z8, z8};
#pragma unroll
    for (int t = 0; t < 2; ++t) {
      const u16* krow = Kb + (size_t)(rowBase + kRow0 + t*16 + l16) * D_ + h*HD_ + hw*8;
#pragma unroll
      for (int dc = 0; dc < 2; ++dc) {
        BFrag bK;
        const u16* p = krow + dc*32;
        bK.q[0] = *(const v4u*)p;
        bK.q[1] = *(const v4u*)(p + 16);
        sfr[t] = wmma_bf16(aQ[dc], bK, sfr[t]);
      }
    }

    // scale + causal mask in C layout (row = qRow0 + hw*8 + r, col = kRow0 + t*16 + l16)
    float sval[2][8];
#pragma unroll
    for (int t = 0; t < 2; ++t)
#pragma unroll
      for (int r = 0; r < 8; ++r) {
        int q = qRow0 + hw*8 + r;
        int k = kRow0 + t*16 + l16;
        float v = sfr[t][r] * 0.125f;               // 1/sqrt(64)
        sval[t][r] = (k > q) ? -__builtin_inff() : v;
      }

    // online softmax: row reductions across the 16 lanes of each half-wave
    float alpha[8], pw[2][8];
#pragma unroll
    for (int r = 0; r < 8; ++r) {
      float v = fmaxf(sval[0][r], sval[1][r]);
#pragma unroll
      for (int off = 1; off < 16; off <<= 1) v = fmaxf(v, __shfl_xor(v, off, 32));
      float mn = fmaxf(m[r], v);
      alpha[r] = __expf(m[r] - mn);
      m[r] = mn;
      pw[0][r] = __expf(sval[0][r] - mn);
      pw[1][r] = __expf(sval[1][r] - mn);
      float rs = pw[0][r] + pw[1][r];
#pragma unroll
      for (int off = 1; off < 16; off <<= 1) rs += __shfl_xor(rs, off, 32);
      lsum[r] = lsum[r] * alpha[r] + rs;
    }
#pragma unroll
    for (int t = 0; t < 4; ++t)
#pragma unroll
      for (int r = 0; r < 8; ++r) o[t][r] *= alpha[r];

    // P: C layout -> LDS -> A-fragment (bf16). Same-wave DS ops are in-order.
#pragma unroll
    for (int t = 0; t < 2; ++t)
#pragma unroll
      for (int r = 0; r < 8; ++r)
        pstage[(hw*8 + r) * 32 + t*16 + l16] = f2bf(pw[t][r]);
    __asm__ volatile("s_wait_dscnt 0" ::: "memory");

    BFrag aP;
    {
      const u16* ps = &pstage[l16 * 32 + hw*8];
      aP.q[0] = *(const v4u*)ps;
      aP.q[1] = *(const v4u*)(ps + 16);
    }

    // O += P @ V  (V stored transposed: contiguous per-lane B-fragment loads)
#pragma unroll
    for (int t = 0; t < 4; ++t) {
      BFrag bV;
      const u16* vp = Vt + (size_t)(h*HD_ + t*16 + l16) * M_TOT + rowBase + kRow0 + hw*8;
      bV.q[0] = *(const v4u*)vp;
      bV.q[1] = *(const v4u*)(vp + 16);
      o[t] = wmma_bf16(aP, bV, o[t]);
    }
    __asm__ volatile("" ::: "memory");  // order pstage reuse across iterations
  }

  // normalize and store combined (bf16)
#pragma unroll
  for (int r = 0; r < 8; ++r) {
    float inv = 1.0f / lsum[r];
    int row = rowBase + qRow0 + hw*8 + r;
#pragma unroll
    for (int t = 0; t < 4; ++t)
      Cmb[(size_t)row * D_ + h*HD_ + t*16 + l16] = f2bf(o[t][r] * inv);
  }
}

// ---------------------------------------------------------------- out GEMM
__global__ void __launch_bounds__(32)
gemm_out_kernel(const u16* __restrict__ Ab, const u16* __restrict__ WoT,
                float* __restrict__ out) {
  __shared__ __align__(16) u16 atile[2][64 * 32];
  const int lane = threadIdx.x;
  const int hw = lane >> 4, l16 = lane & 15;
  const int mBase = blockIdx.x * 64;
  const int nBase = blockIdx.y * 64;

  v8f acc[4][4];
  gemm64_compute(Ab, WoT, mBase, nBase, lane, &atile[0][0], &atile[1][0], acc);

#pragma unroll
  for (int s = 0; s < 4; ++s)
#pragma unroll
    for (int t = 0; t < 4; ++t)
#pragma unroll
      for (int r = 0; r < 8; ++r) {
        int row = mBase + s*16 + hw*8 + r;
        int col = nBase + t*16 + l16;
        out[(size_t)row * D_ + col] = acc[s][t][r];
      }
}

// ---------------------------------------------------------------- launch
extern "C" void kernel_launch(void* const* d_in, const int* in_sizes, int n_in,
                              void* d_out, int out_size, void* d_ws, size_t ws_size,
                              hipStream_t stream) {
  const float* x  = (const float*)d_in[0];
  const float* Wq = (const float*)d_in[1];
  const float* Wk = (const float*)d_in[2];
  const float* Wv = (const float*)d_in[3];
  const float* Wo = (const float*)d_in[4];

  // workspace layout (bf16 elements), ~50 MB total
  u16* xb  = (u16*)d_ws;                       // [4096 x 1024]
  u16* WqT = xb  + (size_t)M_TOT * D_;         // [1024 x 1024] transposed
  u16* WkT = WqT + (size_t)D_ * D_;
  u16* WvT = WkT + (size_t)D_ * D_;
  u16* WoT = WvT + (size_t)D_ * D_;
  u16* Qb  = WoT + (size_t)D_ * D_;            // [4096 x 1024]
  u16* Kb  = Qb  + (size_t)M_TOT * D_;         // [4096 x 1024]
  u16* Vt  = Kb  + (size_t)M_TOT * D_;         // [1024 x 4096] (transposed V)
  u16* Cmb = Vt  + (size_t)M_TOT * D_;         // [4096 x 1024]

  cvt_x_kernel<<<(M_TOT * D_ + 255) / 256, 256, 0, stream>>>(x, xb, M_TOT * D_);
  cvt_wt_kernel<<<(D_ * D_) / 256, 256, 0, stream>>>(Wq, WqT);
  cvt_wt_kernel<<<(D_ * D_) / 256, 256, 0, stream>>>(Wk, WkT);
  cvt_wt_kernel<<<(D_ * D_) / 256, 256, 0, stream>>>(Wv, WvT);
  cvt_wt_kernel<<<(D_ * D_) / 256, 256, 0, stream>>>(Wo, WoT);

  gemm_qkv_kernel<<<dim3(M_TOT / 64, D_ / 64, 3), 32, 0, stream>>>(
      xb, WqT, WkT, WvT, Qb, Kb, Vt);

  attn_kernel<<<dim3(S_ / 16, H_, B_), 32, 0, stream>>>(Qb, Kb, Vt, Cmb);

  gemm_out_kernel<<<dim3(M_TOT / 64, D_ / 64), 32, 0, stream>>>(
      Cmb, WoT, (float*)d_out);
}
